// FingerprintModule_23871428231902
// MI455X (gfx1250) — matrix operand order repaired and checked by reading, not compiled, not run on hardware
//
#include <hip/hip_runtime.h>
#include <math.h>

// Problem constants (from reference): B=4096, T=64, D=256, E=128, H=256, 4H=1024.
constexpr int BB = 4096;
constexpr int TT = 64;
constexpr int DD = 256;
constexpr int EE = 128;
constexpr int HH = 256;
constexpr int GG = 4 * HH;

typedef __attribute__((ext_vector_type(16))) __bf16        v16bf;
typedef __attribute__((ext_vector_type(8)))  float         v8f;
typedef __attribute__((ext_vector_type(4)))  unsigned int  v4u;
typedef __attribute__((ext_vector_type(4)))  float         v4f;

union Frag { v16bf v; v4u q[2]; };
union Acc  { v8f v; float f[8]; };

__device__ __forceinline__ float sigm(float x) { return 1.0f / (1.0f + __expf(-x)); }

// ---------------------------------------------------------------------------
// Prep: convert W_enc [E,D], W_ih [4H,E], W_hh [4H,H] fp32 -> bf16 (row major).
// Row-major bf16 is already the ideal WMMA B-fragment source: lane l of a
// fragment reads 8+8 bf16 at row (nt*16 + l%16), cols kt*32+(l/16)*8 + {0..7, 16..23}.
// ---------------------------------------------------------------------------
__global__ void prep_weights_bf16(const float* __restrict__ we,
                                  const float* __restrict__ wi,
                                  const float* __restrict__ wh,
                                  __bf16* __restrict__ o) {
  const int i  = blockIdx.x * blockDim.x + threadIdx.x;
  const int n0 = EE * DD;
  const int n1 = n0 + GG * EE;
  const int n2 = n1 + GG * HH;
  if (i < n0)       o[i] = (__bf16)we[i];
  else if (i < n1)  o[i] = (__bf16)wi[i - n0];
  else if (i < n2)  o[i] = (__bf16)wh[i - n1];
}

// ---------------------------------------------------------------------------
// Fully fused encoder + LSTM + loss kernel.
// Grid: B/16 workgroups of 256 threads (8 wave32). Each workgroup owns 16
// batch rows and walks t = 0..T-1; the recurrence is private to a batch row,
// so no inter-workgroup sync is needed.
// ---------------------------------------------------------------------------
__global__ __launch_bounds__(256) void lstm_fp_fused(
    const float* __restrict__ x,
    const __bf16* __restrict__ wbf,
    const float* __restrict__ b_enc,
    const float* __restrict__ b_ih,
    const float* __restrict__ b_hh,
    float* __restrict__ out) {
  __shared__ __bf16 lds_enc[16 * EE];  // 4 KB bf16 enc tile
  __shared__ __bf16 lds_h[16 * HH];    // 8 KB bf16 hidden state
  __shared__ float  lds_part[8 * 16];  // per-wave per-row sum(diff^2)

  const int tid = threadIdx.x;
  const int w   = tid >> 5;   // wave 0..7
  const int l   = tid & 31;   // lane
  const int ln  = l & 15;     // lane % 16
  const int hl  = l >> 4;     // lane / 16
  const int b0  = blockIdx.x * 16;

  float* out_hn   = out;                           // [B,H]
  float* out_loss = out + (size_t)BB * HH;         // [B]
  float* out_ldim = out_loss + BB;                 // [B,H]
  float* out_ldt  = out_ldim + (size_t)BB * HH;    // [B,T,H]

  for (int i = tid; i < 16 * HH; i += 256) lds_h[i] = (__bf16)0.0f;  // h0 = 0
  __syncthreads();

  // Hoisted biases (constant per lane across all timesteps).
  const float benc_l = b_enc[w * 16 + ln];
  float bias_g[8];
#pragma unroll
  for (int g = 0; g < 4; ++g)
#pragma unroll
    for (int n = 0; n < 2; ++n) {
      const int col = g * HH + w * 32 + n * 16 + ln;
      bias_g[g * 2 + n] = b_ih[col] + b_hh[col];
    }

  Acc c_st[2], ld_acc[2];
#pragma unroll
  for (int n = 0; n < 2; ++n)
#pragma unroll
    for (int r = 0; r < 8; ++r) { c_st[n].f[r] = 0.f; ld_acc[n].f[r] = 0.f; }
  float loss_acc = 0.f;

  const float* xrow = x + (size_t)(b0 + ln) * TT * DD;  // A-frag row for this lane

#pragma unroll 1
  for (int t = 0; t < TT; ++t) {
    // Opaque zero offset per iteration: defeats LICM of the 104 t-invariant
    // weight B-fragments (which would spill to scratch) while PRESERVING the
    // global address-space provenance of wbf -> plain global_load_b128, not
    // flat_load (flat would burn DScnt + the LDS path on every weight read).
    size_t wofs = 0;
    asm volatile("" : "+s"(wofs));
    const __bf16* Wenc = wbf + wofs;
    const __bf16* Wih  = wbf + wofs + EE * DD;
    const __bf16* Whh  = wbf + wofs + EE * DD + GG * EE;

    //================ Stage A: enc = tanh(x_t * W_enc^T + b_enc) ================
    // Wave w computes enc columns [16w, 16w+16); K = D = 256 -> 8 WMMA.
    // Two accumulators break the WMMA->WMMA D->C RAW hazard chain.
    Acc ea0, ea1;
#pragma unroll
    for (int r = 0; r < 8; ++r) { ea0.f[r] = 0.f; ea1.f[r] = 0.f; }
    const float* xr = xrow + (size_t)t * DD;
#pragma unroll
    for (int kt = 0; kt < 8; ++kt) {
      // A fragment: lane holds K = kt*32 + hl*8 + {0..7} and +16..+23 (fp32 -> bf16)
      const int off = kt * 32 + hl * 8;
      const v4f f0 = *(const v4f*)(xr + off);
      const v4f f1 = *(const v4f*)(xr + off + 4);
      const v4f f2 = *(const v4f*)(xr + off + 16);
      const v4f f3 = *(const v4f*)(xr + off + 20);
      Frag a;
      a.v[0]  = (__bf16)f0.x; a.v[1]  = (__bf16)f0.y; a.v[2]  = (__bf16)f0.z; a.v[3]  = (__bf16)f0.w;
      a.v[4]  = (__bf16)f1.x; a.v[5]  = (__bf16)f1.y; a.v[6]  = (__bf16)f1.z; a.v[7]  = (__bf16)f1.w;
      a.v[8]  = (__bf16)f2.x; a.v[9]  = (__bf16)f2.y; a.v[10] = (__bf16)f2.z; a.v[11] = (__bf16)f2.w;
      a.v[12] = (__bf16)f3.x; a.v[13] = (__bf16)f3.y; a.v[14] = (__bf16)f3.z; a.v[15] = (__bf16)f3.w;
      // B fragment straight from row-major bf16 W_enc (L2 resident).
      Frag b;
      const __bf16* bp = Wenc + (size_t)(w * 16 + ln) * DD + kt * 32 + hl * 8;
      b.q[0] = *(const v4u*)bp;
      b.q[1] = *(const v4u*)(bp + 16);
      if (kt & 1)
        ea1.v = __builtin_amdgcn_wmma_f32_16x16x32_bf16(false, a.v, false, b.v,
                                                        (short)0, ea1.v, false, false);
      else
        ea0.v = __builtin_amdgcn_wmma_f32_16x16x32_bf16(false, a.v, false, b.v,
                                                        (short)0, ea0.v, false, false);
    }
#pragma unroll
    for (int r = 0; r < 8; ++r) {
      const float e = tanhf(ea0.f[r] + ea1.f[r] + benc_l);
      const int M = r + hl * 8;                       // C layout: M = r + 8*(lane/16)
      lds_enc[M * EE + w * 16 + ln] = (__bf16)e;
    }
    if (t + 1 < TT) __builtin_prefetch(xr + DD, 0, 0);  // next x row -> global_prefetch_b8
    __syncthreads();

    //================ Stage B: gates = enc*W_ih^T + h*W_hh^T + b ================
    // Wave w owns hidden columns [32w, 32w+32): its own i/f/g/o tiles, so the
    // pointwise LSTM update never leaves registers.
    // K-outer loop: exactly ONE A-fragment live at a time; the 8 rotating
    // accumulators also hide the WMMA->WMMA hazard window.
    Acc gacc[8];
#pragma unroll
    for (int a = 0; a < 8; ++a)
#pragma unroll
      for (int r = 0; r < 8; ++r) gacc[a].f[r] = 0.f;

#pragma unroll
    for (int kt = 0; kt < 4; ++kt) {               // K over E = 128 (enc term)
      Frag a;
      const __bf16* p = lds_enc + ln * EE + kt * 32 + hl * 8;
      a.q[0] = *(const v4u*)p;
      a.q[1] = *(const v4u*)(p + 16);
#pragma unroll
      for (int g = 0; g < 4; ++g)
#pragma unroll
        for (int n = 0; n < 2; ++n) {
          const int ai = g * 2 + n;
          Frag b;
          const __bf16* bp =
              Wih + (size_t)(g * HH + w * 32 + n * 16 + ln) * EE + kt * 32 + hl * 8;
          b.q[0] = *(const v4u*)bp;
          b.q[1] = *(const v4u*)(bp + 16);
          gacc[ai].v = __builtin_amdgcn_wmma_f32_16x16x32_bf16(false, a.v, false, b.v,
                                                               (short)0, gacc[ai].v, false, false);
        }
    }
#pragma unroll
    for (int kt = 0; kt < 8; ++kt) {               // K over H = 256 (recurrent term)
      Frag a;
      const __bf16* p = lds_h + ln * HH + kt * 32 + hl * 8;
      a.q[0] = *(const v4u*)p;
      a.q[1] = *(const v4u*)(p + 16);
#pragma unroll
      for (int g = 0; g < 4; ++g)
#pragma unroll
        for (int n = 0; n < 2; ++n) {
          const int ai = g * 2 + n;
          Frag b;
          const __bf16* bp =
              Whh + (size_t)(g * HH + w * 32 + n * 16 + ln) * HH + kt * 32 + hl * 8;
          b.q[0] = *(const v4u*)bp;
          b.q[1] = *(const v4u*)(bp + 16);
          gacc[ai].v = __builtin_amdgcn_wmma_f32_16x16x32_bf16(false, a.v, false, b.v,
                                                               (short)0, gacc[ai].v, false, false);
        }
    }
    __syncthreads();  // all waves done reading lds_h / lds_enc before h is rewritten

    //================ Activations, state update, fused losses ================
    const bool last = (t == TT - 1);
    float part[8];
#pragma unroll
    for (int r = 0; r < 8; ++r) part[r] = 0.f;
#pragma unroll
    for (int n = 0; n < 2; ++n) {
#pragma unroll
      for (int r = 0; r < 8; ++r) {
        const float ig = sigm (gacc[0 + n].f[r] + bias_g[0 + n]);
        const float fg = sigm (gacc[2 + n].f[r] + bias_g[2 + n]);
        const float gg = tanhf(gacc[4 + n].f[r] + bias_g[4 + n]);
        const float og = sigm (gacc[6 + n].f[r] + bias_g[6 + n]);
        const float c  = fg * c_st[n].f[r] + ig * gg;
        c_st[n].f[r] = c;
        const float h = og * tanhf(c);
        const int M   = r + hl * 8;
        const int col = w * 32 + n * 16 + ln;
        lds_h[M * HH + col] = (__bf16)h;
        const size_t row = (size_t)(b0 + M);
        if (!last) {
          const float d  = h - x[(row * TT + (t + 1)) * DD + col];
          const float ad = fabsf(d);
          ld_acc[n].f[r] += ad;
          out_ldt[(row * TT + t) * HH + col] = ad;
          part[r] += d * d;
        } else {
          out_ldt[(row * TT + t) * HH + col] = 0.f;   // zero-pad last slice
          out_hn[row * HH + col] = h;                 // h_n
        }
      }
    }

    // Deterministic per-row sum(diff^2): butterfly over the 16 lanes that share
    // a row, then fixed-order cross-wave sum via LDS (no atomics).
    if (!last) {
#pragma unroll
      for (int r = 0; r < 8; ++r) {
        float v = part[r];
        v += __shfl_xor(v, 1);
        v += __shfl_xor(v, 2);
        v += __shfl_xor(v, 4);
        v += __shfl_xor(v, 8);
        part[r] = v;
      }
      if (ln == 0) {
#pragma unroll
        for (int r = 0; r < 8; ++r) lds_part[w * 16 + hl * 8 + r] = part[r];
      }
    }
    __syncthreads();
    if (!last && tid < 16) {
      float s = 0.f;
#pragma unroll
      for (int wv = 0; wv < 8; ++wv) s += lds_part[wv * 16 + tid];
      loss_acc += sqrtf(s);
    }
  }

  //================ Epilogue: loss_dim and loss ================
  const float inv = 1.0f / (float)(TT - 1);
#pragma unroll
  for (int n = 0; n < 2; ++n)
#pragma unroll
    for (int r = 0; r < 8; ++r) {
      const int M   = r + hl * 8;
      const int col = w * 32 + n * 16 + ln;
      out_ldim[(size_t)(b0 + M) * HH + col] = ld_acc[n].f[r] * inv;
    }
  if (tid < 16) out_loss[b0 + tid] = loss_acc * inv;
}

// ---------------------------------------------------------------------------
extern "C" void kernel_launch(void* const* d_in, const int* in_sizes, int n_in,
                              void* d_out, int out_size, void* d_ws, size_t ws_size,
                              hipStream_t stream) {
  (void)in_sizes; (void)n_in; (void)out_size; (void)ws_size;
  const float* x     = (const float*)d_in[0];
  const float* W_enc = (const float*)d_in[1];
  const float* b_enc = (const float*)d_in[2];
  const float* W_ih  = (const float*)d_in[3];
  const float* b_ih  = (const float*)d_in[4];
  const float* W_hh  = (const float*)d_in[5];
  const float* b_hh  = (const float*)d_in[6];
  __bf16* wbf = (__bf16*)d_ws;   // 832 KB of bf16 weights in scratch
  float* out = (float*)d_out;

  const int total = EE * DD + GG * EE + GG * HH;
  prep_weights_bf16<<<(total + 255) / 256, 256, 0, stream>>>(W_enc, W_ih, W_hh, wbf);
  lstm_fp_fused<<<BB / 16, 256, 0, stream>>>(x, wbf, b_enc, b_ih, b_hh, out);
}
